// SpixConvDenoiser_48584670052566
// MI455X (gfx1250) — matrix operand order repaired and checked by reading, not compiled, not run on hardware
//
#include <hip/hip_runtime.h>

// ---------------------------------------------------------------------------
// SpixConvDenoiser fused kernel for gfx1250 (MI455X), wave32 + WMMA bf16.
//
// Stage 1 (rweight): Gram matrix between 16 center pixels and the 22-wide
//   neighbor row for each dy, via v_wmma_f32_16x16x32_bf16 (2 N-tiles x 4
//   K-chunks per dy, S=128).
// Stage 2 (conv): u[m][k*6+i] = rweight[k]*patch[i,k] (bf16 in LDS), then
//   out(16x16) = u(16x320) x Wt(320x16) as 10 chained WMMAs.
// ---------------------------------------------------------------------------

typedef __attribute__((ext_vector_type(16))) __bf16 v16bf;
typedef __attribute__((ext_vector_type(8)))  __bf16 v8bf;
typedef __attribute__((ext_vector_type(2)))  __bf16 v2bf;
typedef __attribute__((ext_vector_type(8)))  float  v8f;

#define NB    4
#define NC    6
#define NS    128
#define NH    320
#define NW    320
#define HW    (NH*NW)
#define KS    7
#define KK    (KS*KS)     // 49
#define CIN   (NC*KK)     // 294
#define TW    16          // tile width (output pixels per wave)
#define CH    32          // rows per workgroup
#define RWD   22          // ring width = TW + 6 (halo 3 each side)
#define SSTR  136         // sims ring s-stride (bf16 elems), multiple of 8
#define USTR  320         // u j-stride (bf16 elems), 10 K-chunks of 32
#define RCOL  24          // rweight scratch columns (>=22)
#define XSTR  8           // xr per-pixel stride (f32), 32B aligned rows

// LDS layout (bytes):
#define OFF_RING 0                         // [7][RWD][SSTR] bf16 = 41888
#define OFF_RWD  41888                     // [16][RCOL]     f32  =  1536
#define OFF_XR   43424                     // [7][RWD][XSTR] f32  =  4928
#define OFF_U    48352                     // [16][USTR]     bf16 = 10240
#define SMEM_BYTES 58592

__device__ __forceinline__ int ring_slot(int hr) { return (hr + 70) % 7; }
__device__ __forceinline__ int iclamp(int v, int lo, int hi) {
  return v < lo ? lo : (v > hi ? hi : v);
}

// Stream one image row (sims f32->bf16 transpose + x channels) into the ring.
__device__ __forceinline__ void load_row(const float* __restrict__ sims,
                                         const float* __restrict__ x,
                                         __bf16* ring, float* xr,
                                         int b, int hr, int w0, int lane) {
  const int  slot  = ring_slot(hr);
  const bool inrow = (hr >= 0) && (hr < NH);   // wave-uniform
  const int  wbase = w0 - 3;

  // sims: 128 s x 22 w = 2816 elems = 11 blocks x 8 x 32 lanes.
  // Walk (s, wl) incrementally: idx += 32  =>  wl += 10, s += 1 (+wrap).
  if (inrow) {
    int wl   = (lane < RWD) ? lane : lane - RWD;
    int gidx = (lane < RWD) ? 0 : HW;                 // s*HW (f32 elems)
    int loff = (slot * RWD + wl) * SSTR + ((lane < RWD) ? 0 : 1);
    const float* grow = sims + (size_t)b * NS * HW + (size_t)hr * NW;
#pragma unroll 1
    for (int blk = 0; blk < 11; ++blk) {
      float vv[8];                                    // 8 loads in flight
      int   lo8[8];
#pragma unroll
      for (int t = 0; t < 8; ++t) {
        int gw  = wbase + wl;
        int gwc = iclamp(gw, 0, NW - 1);
        float v = grow[gidx + gwc];
        vv[t]  = (gw == gwc) ? v : 0.0f;              // width zero-padding
        lo8[t] = loff;
        bool wrap = (wl + 10) >= RWD;
        wl   += wrap ? (10 - RWD) : 10;
        gidx += wrap ? (2 * HW) : HW;
        loff += wrap ? (1 + (10 - RWD) * SSTR) : (1 + 10 * SSTR);
      }
#pragma unroll
      for (int t = 0; t < 8; ++t)
        ring[lo8[t]] = (__bf16)vv[t];
    }
  } else {
    // height zero-padding: clear the whole ring row with 16B stores
    v8bf z = {};
    v8bf* rrow = (v8bf*)(ring + (size_t)slot * RWD * SSTR);
#pragma unroll 1
    for (int i = lane; i < (RWD * SSTR) / 8; i += 32)
      rrow[i] = z;
  }

  // x: 22 pixels x 6 channels; lane = pixel, 6 contiguous f32 per 32B row.
  if (lane < RWD) {
    int gw  = wbase + lane;
    int gwc = iclamp(gw, 0, NW - 1);
    float* xp = xr + (slot * RWD + lane) * XSTR;
    const float* gx = x + (size_t)b * NC * HW + (size_t)hr * NW;
#pragma unroll
    for (int i = 0; i < NC; ++i) {
      float v = 0.0f;
      if (inrow) { v = gx[i * HW + gwc]; if (gw != gwc) v = 0.0f; }
      xp[i] = v;
    }
  }
}

__global__ __launch_bounds__(32)
void spixconv_wmma_kernel(const float* __restrict__ x,
                          const float* __restrict__ sims,
                          const float* __restrict__ Wg,
                          const float* __restrict__ bias,
                          float* __restrict__ out) {
  extern __shared__ char smem[];
  __bf16* ring = (__bf16*)(smem + OFF_RING);
  float*  rwd  = (float*) (smem + OFF_RWD);
  float*  xr   = (float*) (smem + OFF_XR);
  __bf16* u    = (__bf16*)(smem + OFF_U);

  const int lane = threadIdx.x;          // 0..31 (wave32)
  const int w0   = blockIdx.x * TW;      // 0..304
  const int h0   = blockIdx.y * CH;      // 0..288
  const int b    = blockIdx.z;           // 0..3
  const int half = lane >> 4;            // 0/1
  const int l16  = lane & 15;

  // ---- one-time: stage permuted W^T (bf16, zero-padded) into the u buffer,
  // then pull the 10 B-fragments out with two 16B LDS reads each.
  // u element j = k*6+i pairs with Wg[o*294 + i*49 + k].
  const int ocl = iclamp(l16, 0, NC - 1);
#pragma unroll 1
  for (int idx = lane; idx < 16 * USTR; idx += 32) {
    int o = idx / USTR, j = idx - o * USTR;
    int jc = iclamp(j, 0, CIN - 1);
    int k = jc / NC, i = jc - k * NC;
    float v = Wg[iclamp(o, 0, NC - 1) * CIN + i * KK + k];
    u[idx] = (__bf16)(((o < NC) && (j < CIN)) ? v : 0.0f);
  }
  // B layout: col N=lane&15; e<8 -> K'=kc*32+half*8+e, e>=8 -> K'=kc*32+16+half*8+(e-8).
  v16bf wfrag[10];
#pragma unroll
  for (int kc = 0; kc < 10; ++kc) {
    const __bf16* wb = u + l16 * USTR + kc * 32 + half * 8;
    v8bf lo = *(const v8bf*)(wb);
    v8bf hi = *(const v8bf*)(wb + 16);
    v16bf f;
#pragma unroll
    for (int e = 0; e < 8; ++e) { f[e] = lo[e]; f[e + 8] = hi[e]; }
    wfrag[kc] = f;
  }
  const float bo = (l16 < NC) ? bias[ocl] : 0.0f;
  // NOTE: u columns [294,320) stay zero forever (scatter writes only j<294);
  // u rows are fully rewritten by pixel data before stage-2 reads them.

  // ---- prologue: rows h0-3 .. h0+2 into the 7-row ring ----
#pragma unroll 1
  for (int hr = h0 - 3; hr <= h0 + 2; ++hr)
    load_row(sims, x, ring, xr, b, hr, w0, lane);
  __syncthreads();

  for (int h = h0; h < h0 + CH; ++h) {
    // stream next ring row (h+3); it is consumed this iteration (dy=6)
    load_row(sims, x, ring, xr, b, h + 3, w0, lane);

    // prefetch next row (h+4): one line per s-plane, 4 per lane
    {
      int hpf = iclamp(h + 4, 0, NH - 1);
      const float* pfb = sims + (((size_t)b * NS + lane) * NH + hpf) * NW
                              + (w0 > 3 ? w0 - 3 : 0);
#pragma unroll
      for (int q = 0; q < 4; ++q)
        __builtin_prefetch(pfb + (size_t)q * 32 * HW, 0, 0);
    }
    __syncthreads();

    // ---- Stage 1: center-row A fragments (16x32 bf16 per K-chunk) ----
    // A layout: lane row m=lane&15; e<8 -> K=kc*32+half*8+e,
    //           e>=8 -> K=kc*32+16+half*8+(e-8).  Two 16B LDS chunks.
    const int cslot = ring_slot(h);
    v16bf afrag[4];
#pragma unroll
    for (int kc = 0; kc < 4; ++kc) {
      const __bf16* ab = ring + (size_t)(cslot * RWD + 3 + l16) * SSTR
                              + kc * 32 + half * 8;
      v8bf lo = *(const v8bf*)(ab);
      v8bf hi = *(const v8bf*)(ab + 16);
      v16bf f;
#pragma unroll
      for (int e = 0; e < 8; ++e) { f[e] = lo[e]; f[e + 8] = hi[e]; }
      afrag[kc] = f;
    }

    for (int dy = 0; dy < KS; ++dy) {
      const int nslot = ring_slot(h + dy - 3);
#pragma unroll
      for (int nt = 0; nt < 2; ++nt) {
        int n = l16 + nt * 16;               // neighbor pixel in ring coords
        if (n > RWD - 1) n = RWD - 1;        // clamp; cols >=22 unused
        v8f acc = {};
#pragma unroll
        for (int kc = 0; kc < 4; ++kc) {
          const __bf16* bb = ring + (size_t)(nslot * RWD + n) * SSTR
                                  + kc * 32 + half * 8;
          v8bf lo = *(const v8bf*)(bb);
          v8bf hi = *(const v8bf*)(bb + 16);
          v16bf bf;
#pragma unroll
          for (int e = 0; e < 8; ++e) { bf[e] = lo[e]; bf[e + 8] = hi[e]; }
          acc = __builtin_amdgcn_wmma_f32_16x16x32_bf16(
              false, afrag[kc], false, bf, (short)0, acc, false, false);
        }
        // D layout: VGPR r -> row r+8*half, col = lane&15 (+16 for tile 1)
        int col = l16 + nt * 16;
        if (col < RCOL) {
#pragma unroll
          for (int r = 0; r < 8; ++r)
            rwd[(r + 8 * half) * RCOL + col] = acc[r];
        }
      }

      // scatter this dy's rweight*patch products into u (DS ops in-order).
      // lane = pixel l16; half splits dx by parity; all offsets immediate.
      {
        const float* xrow = xr + (nslot * RWD + l16) * XSTR;  // + dx*XSTR
        const float* rrow = rwd + l16 * RCOL + l16;           // + dx
        __bf16* up = u + l16 * USTR + dy * 42;                // + dx*6
#pragma unroll
        for (int it = 0; it < 4; ++it) {
          int dx = 2 * it + half;
          if (dx < KS) {
            float rv = rrow[dx];
            const float* xp = xrow + dx * XSTR;
#pragma unroll
            for (int pr = 0; pr < 3; ++pr) {
              v2bf q;
              q[0] = (__bf16)(rv * xp[2 * pr]);
              q[1] = (__bf16)(rv * xp[2 * pr + 1]);
              *(v2bf*)(up + dx * NC + 2 * pr) = q;
            }
          }
        }
      }
    }

    // ---- Stage 2: out(16x16) = u(16x320) x Wt(320x16), 10 chained WMMAs ----
    v8f acc2 = {};
#pragma unroll
    for (int kc = 0; kc < 10; ++kc) {
      const __bf16* ub = u + l16 * USTR + kc * 32 + half * 8;
      v8bf lo = *(const v8bf*)(ub);
      v8bf hi = *(const v8bf*)(ub + 16);
      v16bf af;
#pragma unroll
      for (int e = 0; e < 8; ++e) { af[e] = lo[e]; af[e + 8] = hi[e]; }
      acc2 = __builtin_amdgcn_wmma_f32_16x16x32_bf16(
          false, af, false, wfrag[kc], (short)0, acc2, false, false);
    }
    // D2[m][o]: lane col o=lane&15, VGPR r -> pixel m=r+8*half -> w contiguous
    if (l16 < NC) {
      size_t base = (((size_t)b * NC + l16) * NH + h) * NW + w0 + 8 * half;
#pragma unroll
      for (int r = 0; r < 8; ++r)
        out[base + r] = acc2[r] + bo;
    }
    __syncthreads();   // protect ring slot (h-3) before next row overwrite
  }
}

extern "C" void kernel_launch(void* const* d_in, const int* in_sizes, int n_in,
                              void* d_out, int out_size, void* d_ws, size_t ws_size,
                              hipStream_t stream) {
  (void)in_sizes; (void)n_in; (void)out_size; (void)d_ws; (void)ws_size;
  const float* x    = (const float*)d_in[0];
  const float* sims = (const float*)d_in[1];
  const float* Wl   = (const float*)d_in[2];
  const float* bias = (const float*)d_in[3];
  float* out = (float*)d_out;

  dim3 grid(NW / TW, NH / CH, NB);   // 20 x 10 x 4 = 800 single-wave workgroups
  dim3 block(32);
  spixconv_wmma_kernel<<<grid, block, SMEM_BYTES, stream>>>(x, sims, Wl, bias, out);
}
